// ChebyConv_2714419331517
// MI455X (gfx1250) — matrix operand order repaired and checked by reading, not compiled, not run on hardware
//
#include <hip/hip_runtime.h>

#define N_NODES 50000
#define N_EDGES 800000
#define BATCH   8
#define CH      64
#define KORD    6
#define MROWS   (BATCH * N_NODES)        // 400000 rows of 64 features

typedef __attribute__((ext_vector_type(16))) __bf16    v16bf;
typedef __attribute__((ext_vector_type(8)))  float     v8f;
typedef __attribute__((ext_vector_type(8)))  unsigned  v8u;
typedef __attribute__((ext_vector_type(4)))  float     f4;

// round-to-nearest-even fp32 -> bf16 bits (in high 16 of rounded word)
__device__ __forceinline__ unsigned bf16_round_word(float f) {
    unsigned u = __float_as_uint(f);
    return u + 0x7FFFu + ((u >> 16) & 1u);
}
// pack two rounded words into one dword: [f1.bf16 | f0.bf16]
__device__ __forceinline__ unsigned bf16_pack(unsigned r0, unsigned r1) {
    return (r0 >> 16) | (r1 & 0xFFFF0000u);
}
__device__ __forceinline__ float bf16_word_f(unsigned r) {
    return __uint_as_float(r & 0xFFFF0000u);
}

// ---------------- CSR build ----------------
__global__ void zero_counts_kernel(int* __restrict__ counts) {
    int i = blockIdx.x * 256 + threadIdx.x;
    if (i < N_NODES) counts[i] = 0;
}

__global__ void hist_kernel(const long long* __restrict__ ei, int* __restrict__ counts) {
    int e = blockIdx.x * 256 + threadIdx.x;
    if (e < N_EDGES) atomicAdd(&counts[(int)ei[e]], 1);  // row 0 = dst (scatter)
}

// single-block exclusive scan of counts[0..N) -> rowptr[0..N], also copy to cursor
__global__ __launch_bounds__(1024) void scan_kernel(const int* __restrict__ counts,
                                                    int* __restrict__ rowptr,
                                                    int* __restrict__ cursor) {
    __shared__ int sbuf[1024];
    __shared__ int srun;
    const int tid = threadIdx.x;
    if (tid == 0) srun = 0;
    __syncthreads();
    for (int base = 0; base < N_NODES; base += 1024) {
        int i = base + tid;
        int c = (i < N_NODES) ? counts[i] : 0;
        sbuf[tid] = c;
        __syncthreads();
        for (int off = 1; off < 1024; off <<= 1) {
            int v = (tid >= off) ? sbuf[tid - off] : 0;
            __syncthreads();
            sbuf[tid] += v;
            __syncthreads();
        }
        int incl = sbuf[tid];
        int run  = srun;
        if (i < N_NODES) {
            int ex = run + incl - c;
            rowptr[i] = ex;
            cursor[i] = ex;
        }
        __syncthreads();
        if (tid == 1023) srun = run + incl;
        __syncthreads();
    }
    if (tid == 0) rowptr[N_NODES] = srun;
}

__global__ void fill_kernel(const long long* __restrict__ ei, const float* __restrict__ ew,
                            int* __restrict__ cursor, int* __restrict__ csr_src,
                            float* __restrict__ csr_w) {
    int e = blockIdx.x * 256 + threadIdx.x;
    if (e >= N_EDGES) return;
    int dst = (int)ei[e];
    int src = (int)ei[N_EDGES + e];   // row 1 = src (gather)
    int pos = atomicAdd(&cursor[dst], 1);
    csr_src[pos] = src;
    csr_w[pos]   = ew[e];
}

// ---------------- W pre-split: per term k, store hi/lo bf16 B-fragments in WMMA layout
// Layout: Whi[k][tc][lane][v] packed pairs, tc = t*2+kc, 2048 dwords per term per array.
// Element 2v (2v+1) of lane's v16bf = W[kc*32 + (lane>>4)*16 + 2v (+1)][t*16 + (lane&15)]
__global__ __launch_bounds__(256) void wsplit_kernel(const float* __restrict__ W,
                                                     unsigned* __restrict__ Whi,
                                                     unsigned* __restrict__ Wlo) {
    const int k = blockIdx.x;                    // 0..KORD-1
    const float* __restrict__ Wk = W + k * CH * CH;
    unsigned* __restrict__ Hk = Whi + k * 2048;
    unsigned* __restrict__ Lk = Wlo + k * 2048;
    for (int o = threadIdx.x; o < 2048; o += 256) {
        int v    = o & 7;
        int lane = (o >> 3) & 31;
        int tc   = o >> 8;                       // t*2 + kc
        int kc   = tc & 1;
        int t    = tc >> 1;
        int half = lane >> 4, l15 = lane & 15;
        int row  = kc * 32 + half * 16 + 2 * v;
        int col  = t * 16 + l15;
        float f0 = Wk[row * CH + col];
        float f1 = Wk[(row + 1) * CH + col];
        unsigned r0 = bf16_round_word(f0), r1 = bf16_round_word(f1);
        Hk[o] = bf16_pack(r0, r1);
        unsigned s0 = bf16_round_word(f0 - bf16_word_f(r0));
        unsigned s1 = bf16_round_word(f1 - bf16_word_f(r1));
        Lk[o] = bf16_pack(s0, s1);
    }
}

// ---------------- SPMM: xout[b,n,:] = initScale*initSrc[b,n,:] + scale * sum_e w_e * xin[b,src_e,:]
__global__ __launch_bounds__(256) void spmm_csr_kernel(
    const int* __restrict__ rowptr, const int* __restrict__ csr_src,
    const float* __restrict__ csr_w,
    const float* __restrict__ xin, const float* __restrict__ initSrc,
    float* __restrict__ xout, float scale, float initScale)
{
    int idx = blockIdx.x * 256 + threadIdx.x;      // N_NODES*16 threads
    if (idx >= N_NODES * 16) return;
    int n  = idx >> 4;
    int c4 = idx & 15;

    const f4* __restrict__ xin4 = (const f4*)xin;
    const f4* __restrict__ ini4 = (const f4*)initSrc;
    f4* __restrict__       out4 = (f4*)xout;

    f4 acc[BATCH];
#pragma unroll
    for (int b = 0; b < BATCH; ++b) {
        if (initScale != 0.0f) {
            f4 v = ini4[((long)b * N_NODES + n) * 16 + c4];
            acc[b] = initScale * v;
        } else {
            acc[b] = (f4)0.0f;
        }
    }

    int e0 = rowptr[n], e1 = rowptr[n + 1];
    for (int e = e0; e < e1; ++e) {
        int   s = csr_src[e];
        float w = scale * csr_w[e];
#pragma unroll
        for (int b = 0; b < BATCH; ++b) {
            f4 v = xin4[((long)b * N_NODES + s) * 16 + c4];
            acc[b] += w * v;
        }
    }
#pragma unroll
    for (int b = 0; b < BATCH; ++b)
        out4[((long)b * N_NODES + n) * 16 + c4] = acc[b];
}

// ---------------- GEMM: out[128x64 tile] (+)= T[128x64] @ W[64x64] via bf16-split WMMA.
// B fragments are preconverted (wsplit_kernel) -> direct 32B coalesced global loads.
__global__ __launch_bounds__(256) void cheb_gemm_kernel(
    const float* __restrict__ T,
    const unsigned* __restrict__ Whi,   // 2048 dwords, WMMA B layout
    const unsigned* __restrict__ Wlo,
    const float* __restrict__ bias, float* __restrict__ out, int addC)
{
    __shared__ float sA[128 * 64];   // 32 KB

    const int  tid = threadIdx.x;
    const long m0  = (long)blockIdx.x * 128;

    {
        const float* __restrict__ Trow = T + m0 * 64;
        for (int i = tid; i < 128 * 64; i += 256) sA[i] = Trow[i];
    }
    __syncthreads();

    const int wave = tid >> 5;
    const int lane = tid & 31;
    const int half = lane >> 4;
    const int l15  = lane & 15;
    const int arow = wave * 16;
    const long gm  = m0 + arow;

    v8f acc[4];
    if (addC) {
#pragma unroll
        for (int t = 0; t < 4; ++t)
#pragma unroll
            for (int r = 0; r < 8; ++r)
                acc[t][r] = out[(gm + r + 8 * half) * 64 + t * 16 + l15];
    } else {
#pragma unroll
        for (int t = 0; t < 4; ++t) {
            float bv = bias[t * 16 + l15];
#pragma unroll
            for (int r = 0; r < 8; ++r) acc[t][r] = bv;
        }
    }

#pragma unroll
    for (int kc = 0; kc < 2; ++kc) {
        // ---- A fragment (16x32 bf16, contiguous k = 16*half + s), packed-int construction
        v8u pa_hi, pa_lo;
#pragma unroll
        for (int v = 0; v < 8; ++v) {
            int g = v >> 2, r = v & 3;
            int k = kc * 32 + half * 16 + v * 2;
            const float* ap = &sA[(arow + l15) * 64 + k];
            float f0 = ap[0], f1 = ap[1];
            unsigned r0 = bf16_round_word(f0), r1 = bf16_round_word(f1);
            pa_hi[v] = bf16_pack(r0, r1);
            unsigned s0 = bf16_round_word(f0 - bf16_word_f(r0));
            unsigned s1 = bf16_round_word(f1 - bf16_word_f(r1));
            pa_lo[v] = bf16_pack(s0, s1);
        }
        v16bf ahi = __builtin_bit_cast(v16bf, pa_hi);
        v16bf alo = __builtin_bit_cast(v16bf, pa_lo);

#pragma unroll
        for (int t = 0; t < 4; ++t) {
            const int fo = ((t * 2 + kc) * 32 + lane) * 8;
            v16bf bhi = *(const v16bf*)(Whi + fo);   // 32B coalesced load
            v16bf blo = *(const v16bf*)(Wlo + fo);
            // fp32-ish accuracy: hi*hi + lo*hi + hi*lo
            acc[t] = __builtin_amdgcn_wmma_f32_16x16x32_bf16(false, ahi, false, bhi,
                                                             (short)0, acc[t], false, false);
            acc[t] = __builtin_amdgcn_wmma_f32_16x16x32_bf16(false, alo, false, bhi,
                                                             (short)0, acc[t], false, false);
            acc[t] = __builtin_amdgcn_wmma_f32_16x16x32_bf16(false, ahi, false, blo,
                                                             (short)0, acc[t], false, false);
        }
    }

#pragma unroll
    for (int t = 0; t < 4; ++t)
#pragma unroll
        for (int r = 0; r < 8; ++r)
            out[(gm + r + 8 * half) * 64 + t * 16 + l15] = acc[t][r];
}

// ---------------- host ----------------
extern "C" void kernel_launch(void* const* d_in, const int* in_sizes, int n_in,
                              void* d_out, int out_size, void* d_ws, size_t ws_size,
                              hipStream_t stream) {
    const float*     x    = (const float*)d_in[0];      // [B, N, 64]
    const long long* ei   = (const long long*)d_in[1];  // [2, E] int64
    const float*     ew   = (const float*)d_in[2];      // [E]
    const float*     W    = (const float*)d_in[3];      // [K, 64, 64]
    const float*     bias = (const float*)d_in[4];      // [64]
    float*           out  = (float*)d_out;              // [B, N, 64]

    const size_t TBYTES = (size_t)MROWS * CH * sizeof(float);  // 102.4 MB
    char*  ws  = (char*)d_ws;
    size_t off = 0;
    auto take = [&](size_t bytes) -> char* {
        char* p = ws + off;
        off += (bytes + 255) & ~(size_t)255;
        return p;
    };
    float*    t0      = (float*)take(TBYTES);
    float*    t1      = (float*)take(TBYTES);
    int*      rowptr  = (int*)take((N_NODES + 1) * sizeof(int));
    int*      cursor  = (int*)take(N_NODES * sizeof(int));
    int*      csr_src = (int*)take(N_EDGES * sizeof(int));
    float*    csr_w   = (float*)take(N_EDGES * sizeof(float));
    unsigned* Whi     = (unsigned*)take(KORD * 2048 * sizeof(unsigned));
    unsigned* Wlo     = (unsigned*)take(KORD * 2048 * sizeof(unsigned));
    (void)ws_size; (void)in_sizes; (void)n_in; (void)out_size;

    const int EB = (N_EDGES + 255) / 256;
    const int NB = (N_NODES + 255) / 256;
    const int SB = (N_NODES * 16 + 255) / 256;   // 3125
    const int GB = MROWS / 128;                  // 3125

    // ---- CSR build + weight pre-split
    zero_counts_kernel<<<NB, 256, 0, stream>>>(cursor);
    hist_kernel<<<EB, 256, 0, stream>>>(ei, cursor);
    scan_kernel<<<1, 1024, 0, stream>>>(cursor, rowptr, cursor);
    fill_kernel<<<EB, 256, 0, stream>>>(ei, ew, cursor, csr_src, csr_w);
    wsplit_kernel<<<KORD, 256, 0, stream>>>(W, Whi, Wlo);

    // ---- k = 0 : out = x @ W0 + bias
    cheb_gemm_kernel<<<GB, 256, 0, stream>>>(x, Whi + 0 * 2048, Wlo + 0 * 2048, bias, out, 0);

    // ---- k = 1 : t0 = L x ; out += t0 @ W1
    spmm_csr_kernel<<<SB, 256, 0, stream>>>(rowptr, csr_src, csr_w, x, x, t0, 1.0f, 0.0f);
    cheb_gemm_kernel<<<GB, 256, 0, stream>>>(t0, Whi + 1 * 2048, Wlo + 1 * 2048, bias, out, 1);

    // ---- k = 2 : t1 = 2 L t0 - x
    spmm_csr_kernel<<<SB, 256, 0, stream>>>(rowptr, csr_src, csr_w, t0, x, t1, 2.0f, -1.0f);
    cheb_gemm_kernel<<<GB, 256, 0, stream>>>(t1, Whi + 2 * 2048, Wlo + 2 * 2048, bias, out, 1);

    // ---- k = 3 : t0 = 2 L t1 - t0   (in place)
    spmm_csr_kernel<<<SB, 256, 0, stream>>>(rowptr, csr_src, csr_w, t1, t0, t0, 2.0f, -1.0f);
    cheb_gemm_kernel<<<GB, 256, 0, stream>>>(t0, Whi + 3 * 2048, Wlo + 3 * 2048, bias, out, 1);

    // ---- k = 4 : t1 = 2 L t0 - t1   (in place)
    spmm_csr_kernel<<<SB, 256, 0, stream>>>(rowptr, csr_src, csr_w, t0, t1, t1, 2.0f, -1.0f);
    cheb_gemm_kernel<<<GB, 256, 0, stream>>>(t1, Whi + 4 * 2048, Wlo + 4 * 2048, bias, out, 1);

    // ---- k = 5 : t0 = 2 L t1 - t0   (in place)
    spmm_csr_kernel<<<SB, 256, 0, stream>>>(rowptr, csr_src, csr_w, t1, t0, t0, 2.0f, -1.0f);
    cheb_gemm_kernel<<<GB, 256, 0, stream>>>(t0, Whi + 5 * 2048, Wlo + 5 * 2048, bias, out, 1);
}
